// GraphAttentionLayer_1382979469643
// MI455X (gfx1250) — matrix operand order repaired and checked by reading, not compiled, not run on hardware
//
#include <hip/hip_runtime.h>

// -----------------------------------------------------------------------------
// GAT layer, fused flash-attention style for MI455X (gfx1250, wave32, WMMA).
//
// Roofline: graph mask stream = 256MB (mandatory) -> ~11.5us @ 23.3 TB/s.
// 34.4 GFLOP of GEMM done as bf16 WMMA (f32 accum) stays under that floor;
// f32 16x16x4 WMMA would be compute-bound. h / h^T (2MB each, bf16) live in
// L2 (192MB); graph is loaded nontemporally so it does not evict them.
//
// Round-2 loop slimming: rowsum(P) via ones-column WMMA (kills 32 shuffles),
// rescale guarded by __any(max changed) (rare), 1-instruction graph prefetch.
// -----------------------------------------------------------------------------

typedef __attribute__((ext_vector_type(16))) __bf16 v16bf;
typedef __attribute__((ext_vector_type(8)))  __bf16 v8bf;
typedef __attribute__((ext_vector_type(8)))  float  v8f;

#define N_NODES 8192
#define C_DIM   256
#define D_DIM   128
#define MASK_VAL (-1e16f)

// ---------------------------------------------------------------------------
// Kernel 1: h[n,d] = dot(inputs[n,:], W[d,:]) in f32 (exact), then round once
// to bf16 and store both row-major (hb) and transposed (ht).
// ---------------------------------------------------------------------------
__global__ __launch_bounds__(128) void gat_h_kernel(const float* __restrict__ in,
                                                    const float* __restrict__ W,
                                                    __bf16* __restrict__ hb,
                                                    __bf16* __restrict__ ht) {
  __shared__ float xrow[C_DIM];
  const int n = blockIdx.x;
  const int d = threadIdx.x;
  xrow[d]       = in[n * C_DIM + d];
  xrow[d + 128] = in[n * C_DIM + 128 + d];
  __syncthreads();
  const float* wrow = W + d * C_DIM;   // nn.Linear weight [out,in], row-contig
  float acc = 0.f;
  #pragma unroll 8
  for (int c = 0; c < C_DIM; c += 4) {
    acc += xrow[c + 0] * wrow[c + 0];
    acc += xrow[c + 1] * wrow[c + 1];
    acc += xrow[c + 2] * wrow[c + 2];
    acc += xrow[c + 3] * wrow[c + 3];
  }
  const __bf16 hv = (__bf16)acc;
  hb[n * D_DIM + d]   = hv;
  ht[d * N_NODES + n] = hv;
}

// ---------------------------------------------------------------------------
// Kernel 2: fused masked-softmax attention. 512 blocks x 128 threads.
// Each wave: 16 query rows, online softmax over 32-key blocks (stride 4),
// 17 v_wmma_f32_16x16x32_bf16 per block iteration. 4-wave LSE merge in LDS.
// ---------------------------------------------------------------------------
__global__ __launch_bounds__(128) void gat_attn_kernel(const __bf16* __restrict__ hb,
                                                       const __bf16* __restrict__ ht,
                                                       const float* __restrict__ graph,
                                                       const float* __restrict__ bias,
                                                       float* __restrict__ out) {
  __shared__ __attribute__((aligned(32))) __bf16 p_lds[4][16][32]; // P transpose bounce
  __shared__ float Ow[4][16][D_DIM];
  __shared__ float mw[4][16];
  __shared__ float lw[4][16];

  const int wave = threadIdx.x >> 5;
  const int lane = threadIdx.x & 31;
  const int lx   = lane & 15;   // column-in-tile / row-in-A
  const int hi   = lane >> 4;   // half-wave select
  const int iblk = blockIdx.x;
  const int qrow = iblk * 16 + lx;

  // --- A operand: 16 query rows x 128 d, as 4 k-chunks of 32 (resident) ----
  // A k order used here (contiguous, matches B): elem e -> K = kc*32 + 16*hi + e
  v16bf Ah[4];
  #pragma unroll
  for (int kc = 0; kc < 4; ++kc) {
    const __bf16* base = hb + qrow * D_DIM + kc * 32 + 16 * hi;
    v8bf lo = *(const v8bf*)(base);
    v8bf h8 = *(const v8bf*)(base + 8);
    #pragma unroll
    for (int e = 0; e < 8; ++e) { Ah[kc][e] = lo[e]; Ah[kc][e + 8] = h8[e]; }
  }

  // all-ones B tile: rowsum(P) = P x ones via one extra WMMA (no shuffles)
  v16bf ones;
  #pragma unroll
  for (int e = 0; e < 16; ++e) ones[e] = (__bf16)1.0f;
  v8f zero8;
  #pragma unroll
  for (int e = 0; e < 8; ++e) zero8[e] = 0.f;

  // --- accumulators: O[dc] is 16x16 C tile (rows r+8*hi, col d=dc*16+lx) ---
  v8f O[8];
  float m8[8], l8[8];
  #pragma unroll
  for (int dc = 0; dc < 8; ++dc) O[dc] = zero8;
  #pragma unroll
  for (int r = 0; r < 8; ++r) { m8[r] = -__builtin_inff(); l8[r] = 0.f; }

  // per-lane prefetch address covering next j-block's 16 rows x 128B of graph
  const float* pf_base = graph + (iblk * 16 + lx) * N_NODES + hi * 16;

  for (int j = wave; j < N_NODES / 32; j += 4) {
    const int k0 = j * 32;

    // prefetch this wave's NEXT graph block (speculative: OOB is dropped)
    __builtin_prefetch(pf_base + k0 + 128, 0, 0);

    // ---- S = h_q . h_k^T : two 16-key tiles, K=128 in 4 chunks (8 WMMA) ---
    v8f S0 = zero8, S1 = zero8;
    #pragma unroll
    for (int kc = 0; kc < 4; ++kc) {
      // B k order used here (contiguous, matches A): elem e -> K = kc*32 + 16*hi + e, col N = lx (key row)
      v16bf B0 = *(const v16bf*)(hb + (k0 + lx) * D_DIM + kc * 32 + 16 * hi);
      v16bf B1 = *(const v16bf*)(hb + (k0 + 16 + lx) * D_DIM + kc * 32 + 16 * hi);
      S0 = __builtin_amdgcn_wmma_f32_16x16x32_bf16(false, Ah[kc], false, B0,
                                                   (short)0, S0, false, false);
      S1 = __builtin_amdgcn_wmma_f32_16x16x32_bf16(false, Ah[kc], false, B1,
                                                   (short)0, S1, false, false);
    }

    // ---- mask (scores*graph; ==0 -> -1e16) + block row-max ---------------
    float s0a[8], s1a[8], mn[8];
    bool chg = false;
    #pragma unroll
    for (int r = 0; r < 8; ++r) {
      const int row = iblk * 16 + r + 8 * hi;
      const float g0 = __builtin_nontemporal_load(graph + row * N_NODES + k0 + lx);
      const float g1 = __builtin_nontemporal_load(graph + row * N_NODES + k0 + 16 + lx);
      float s0 = S0[r] * g0; s0 = (s0 == 0.f) ? MASK_VAL : s0;
      float s1 = S1[r] * g1; s1 = (s1 == 0.f) ? MASK_VAL : s1;
      // row max across the 16-lane half-wave group
      float t = fmaxf(s0, s1);
      t = fmaxf(t, __shfl_xor(t, 1));
      t = fmaxf(t, __shfl_xor(t, 2));
      t = fmaxf(t, __shfl_xor(t, 4));
      t = fmaxf(t, __shfl_xor(t, 8));
      mn[r] = fmaxf(m8[r], t);
      chg = chg || (mn[r] > m8[r]);
      s0a[r] = s0; s1a[r] = s1;
    }

    // ---- rescale only when a running max actually moved (rare) -----------
    if (__any((int)chg)) {
      #pragma unroll
      for (int r = 0; r < 8; ++r) {
        const float sc = __expf(m8[r] - mn[r]);  // exp(-inf-finite)=0 on 1st
        l8[r] *= sc;
        m8[r] = mn[r];
        #pragma unroll
        for (int dc = 0; dc < 8; ++dc) O[dc][r] *= sc;
      }
    }

    // ---- P = exp(S - m), transpose (C layout) -> A layout via LDS --------
    #pragma unroll
    for (int r = 0; r < 8; ++r) {
      const float P0 = __expf(s0a[r] - m8[r]);
      const float P1 = __expf(s1a[r] - m8[r]);
      p_lds[wave][r + 8 * hi][lx]      = (__bf16)P0;
      p_lds[wave][r + 8 * hi][16 + lx] = (__bf16)P1;
    }
    v16bf Ap;
    {
      const __bf16* pb = &p_lds[wave][lx][0];
      v8bf lo = *(const v8bf*)(pb + 16 * hi);
      v8bf h8 = *(const v8bf*)(pb + 16 * hi + 8);
      #pragma unroll
      for (int e = 0; e < 8; ++e) { Ap[e] = lo[e]; Ap[e + 8] = h8[e]; }
    }

    // ---- l += rowsum(P): P x ones, C=0; every column identical -> no shfl -
    v8f Z = __builtin_amdgcn_wmma_f32_16x16x32_bf16(false, Ap, false, ones,
                                                    (short)0, zero8, false, false);
    #pragma unroll
    for (int r = 0; r < 8; ++r) l8[r] += Z[r];

    // ---- O += P . h_keys : 8 WMMA over d-chunks, B from transposed h -----
    #pragma unroll
    for (int dc = 0; dc < 8; ++dc) {
      // B[K=key 0..31][N=d]: lane lx holds h^T row (dc*16+lx), contiguous keys
      v16bf Bh = *(const v16bf*)(ht + (dc * 16 + lx) * N_NODES + k0 + 16 * hi);
      O[dc] = __builtin_amdgcn_wmma_f32_16x16x32_bf16(false, Ap, false, Bh,
                                                      (short)0, O[dc], false, false);
    }
  }

  // ---- write per-wave partial state, merge 4 waves with LSE combine -------
  #pragma unroll
  for (int dc = 0; dc < 8; ++dc)
    #pragma unroll
    for (int r = 0; r < 8; ++r)
      Ow[wave][r + 8 * hi][dc * 16 + lx] = O[dc][r];
  if (lx == 0) {
    #pragma unroll
    for (int r = 0; r < 8; ++r) {
      mw[wave][8 * hi + r] = m8[r];
      lw[wave][8 * hi + r] = l8[r];
    }
  }
  __syncthreads();

  const int d = threadIdx.x;   // one output column per thread
  for (int row = 0; row < 16; ++row) {
    const float mstar = fmaxf(fmaxf(mw[0][row], mw[1][row]),
                              fmaxf(mw[2][row], mw[3][row]));
    float lsum = 0.f, val = 0.f;
    #pragma unroll
    for (int w = 0; w < 4; ++w) {
      const float s = __expf(mw[w][row] - mstar);
      lsum += lw[w][row] * s;
      val  += Ow[w][row][d] * s;
    }
    out[(iblk * 16 + row) * D_DIM + d] = val / lsum + bias[d];
  }
}

// ---------------------------------------------------------------------------
extern "C" void kernel_launch(void* const* d_in, const int* in_sizes, int n_in,
                              void* d_out, int out_size, void* d_ws, size_t ws_size,
                              hipStream_t stream) {
  const float* inputs = (const float*)d_in[0];   // [1,8192,256]
  const float* graph  = (const float*)d_in[1];   // [8192,8192]
  const float* W      = (const float*)d_in[2];   // [128,256]
  const float* b      = (const float*)d_in[3];   // [128]
  float* out = (float*)d_out;                    // [1,8192,128]

  __bf16* hb = (__bf16*)d_ws;                        // 8192x128 bf16 (2MB)
  __bf16* ht = hb + (size_t)N_NODES * D_DIM;         // 128x8192 bf16 (2MB)

  gat_h_kernel<<<N_NODES, 128, 0, stream>>>(inputs, W, hb, ht);
  gat_attn_kernel<<<N_NODES / 16, 128, 0, stream>>>(hb, ht, graph, b, out);
}
